// SDXLMultiTokenConditioner_50096498540844
// MI455X (gfx1250) — matrix operand order repaired and checked by reading, not compiled, not run on hardware
//
#include <hip/hip_runtime.h>
#include <hip/hip_bf16.h>
#include <cstdint>

#define BDIM 256

typedef __attribute__((ext_vector_type(16))) __bf16 v16bf;
typedef __attribute__((ext_vector_type(8)))  __bf16 v8bf;
typedef __attribute__((ext_vector_type(8)))  float  v8f;

static constexpr int B_N   = 8;
static constexpr int SEQ_N = 16;
static constexpr int IC_N  = 704;
static constexpr int T_N   = 4;
static constexpr int L_N   = 3;
static constexpr int S_N   = 4096;
static constexpr int H1_N  = 768;
static constexpr int H2_N  = 1280;
static constexpr float EPS_F = 1e-5f;

__device__ __forceinline__ float silu_f(float x) {
  return x / (1.0f + __expf(-x));
}

__device__ __forceinline__ v8f wmma_bf16f32(v16bf a, v16bf b, v8f c) {
  // (neg_a, A, neg_b, B, c_mod, C, reuse_a, reuse_b)
  return __builtin_amdgcn_wmma_f32_16x16x32_bf16(false, a, false, b, (short)0, c,
                                                 false, false);
}

__device__ __forceinline__ v16bf pack16_bf(const float* v) {
  v16bf r;
#pragma unroll
  for (int i = 0; i < 16; ++i) r[i] = (__bf16)v[i];
  return r;
}

// ---------------------------------------------------------------------------
// Bulk pass-through copy via the CDNA5 async Global<->LDS engine.
// Each lane shuttles 16B per iteration through its private LDS slot:
//   global_load_async_to_lds_b128  ->  s_wait_asynccnt  ->
//   global_store_async_from_lds_b128 -> s_wait_asynccnt
// GVS addressing: SGPR64 base + 32-bit VGPR byte offset (regions < 2^31 B).
// ---------------------------------------------------------------------------
__global__ __launch_bounds__(BDIM) void copy_async_kernel(
    const float4* __restrict__ src, float4* __restrict__ dst, long n) {
  __shared__ float4 buf[BDIM];  // sole LDS object -> starts at LDS offset 0
  const uint32_t lds_off = (uint32_t)(threadIdx.x * sizeof(float4));
  const long stride = (long)gridDim.x * BDIM;
  for (long i = (long)blockIdx.x * BDIM + threadIdx.x; i < n; i += stride) {
    const uint32_t goff = (uint32_t)(i * sizeof(float4));
    asm volatile("global_load_async_to_lds_b128 %0, %1, %2"
                 :
                 : "v"(lds_off), "v"(goff), "s"(src)
                 : "memory");
    asm volatile("s_wait_asynccnt 0x0" ::: "memory");
    asm volatile("global_store_async_from_lds_b128 %0, %1, %2"
                 :
                 : "v"(goff), "v"(lds_off), "s"(dst)
                 : "memory");
    asm volatile("s_wait_asynccnt 0x0" ::: "memory");  // LDS slot reuse
  }
  __builtin_amdgcn_s_wait_tensorcnt(0);  // TENSORcnt==0: free drain
  // keep `buf` allocated (never provably reachable):
  if ((int)blockIdx.x < 0) dst[0] = buf[threadIdx.x];
}

// Small pass-through copy (float4 grid-stride) for ehs2.
__global__ void copy4_kernel(const float4* __restrict__ s, float4* __restrict__ d,
                             long n) {
  long i = (long)blockIdx.x * blockDim.x + threadIdx.x;
  const long stride = (long)gridDim.x * blockDim.x;
  for (; i < n; i += stride) d[i] = s[i];
}

// ---------------------------------------------------------------------------
// npos[b] = #rows s with sum(|extra[b,s,-256:]|) > 1e-6
// ---------------------------------------------------------------------------
__global__ void npos_kernel(const float* __restrict__ extra, int* __restrict__ npos) {
  const int b = blockIdx.x;
  const int tid = threadIdx.x;  // 256 threads, one per tail channel
  __shared__ float red[BDIM];
  int cnt = 0;
  for (int s = 0; s < SEQ_N; ++s) {
    red[tid] = fabsf(extra[((size_t)b * SEQ_N + s) * IC_N + (IC_N - 256) + tid]);
    __syncthreads();
#pragma unroll
    for (int off = BDIM / 2; off > 0; off >>= 1) {
      if (tid < off) red[tid] += red[tid + off];
      __syncthreads();
    }
    if (tid == 0 && red[0] > 1e-6f) cnt++;
    __syncthreads();
  }
  if (tid == 0) npos[b] = cnt;
}

// ---------------------------------------------------------------------------
// Fused ResMLP (fc1 + 3 GN/residual blocks + out-proj) with WMMA bf16,
// scatter of the out-projection folded into the epilogue.
// One workgroup = one (param-set t, 16-row tile). 8 waves, NT n-tiles/wave.
// ---------------------------------------------------------------------------
template <int HID>
__global__ __launch_bounds__(BDIM) void resmlp_wmma_kernel(
    const float* __restrict__ x,                                   // [128,704]
    const float* __restrict__ fc1_w, const float* __restrict__ fc1_b,
    const float* __restrict__ gn_g,  const float* __restrict__ gn_b,
    const float* __restrict__ blk_w, const float* __restrict__ blk_b,
    const float* __restrict__ out_w, const float* __restrict__ out_b,
    const int*   __restrict__ npos,
    float* __restrict__ out)                                       // [B,S,HID]
{
  static_assert(HID % 128 == 0, "HID must be a multiple of 128");
  constexpr int GS     = HID / 32;   // group-norm group size
  constexpr int NT     = HID / 128;  // n-tiles per wave (8 waves * NT * 16 = HID)
  constexpr int KS_BLK = HID / 32;
  constexpr int KS_FC1 = IC_N / 32;  // 22

  __shared__ __bf16 hsh[16 * HID];   // residual stream, bf16
  __shared__ float  smean[16 * 32];
  __shared__ float  srstd[16 * 32];

  const int t    = blockIdx.y;
  const int r0   = blockIdx.x * 16;
  const int tid  = threadIdx.x;
  const int lane = tid & 31;
  const int wave = tid >> 5;
  // WMMA A layout (16x32 bf16): lane -> row M, K runs [akoff,akoff+8) & +16
  const int am    = lane & 15;
  const int akoff = (lane >> 4) * 8;
  // WMMA B layout (32x16 bf16): lane -> col N, K runs [bkoff,bkoff+8) & +16
  const int bn    = lane & 15;
  const int bkoff = (lane >> 4) * 8;
  // WMMA C/D layout: VGPR j -> M = mbase + j, N = lane&15
  const int mbase = (lane >> 4) * 8;

  v8f acc[NT];

  // ---------------- fc1: h = silu(x @ fc1_w.T + fc1_b) ----------------
  {
    const float* W = fc1_w + (size_t)t * HID * IC_N;
#pragma unroll
    for (int nt = 0; nt < NT; ++nt)
#pragma unroll
      for (int j = 0; j < 8; ++j) acc[nt][j] = 0.0f;

    for (int ks = 0; ks < KS_FC1; ++ks) {
      const float* xr = x + (size_t)(r0 + am) * IC_N + ks * 32 + akoff;
      float av[16];
      *(float4*)&av[0]  = *(const float4*)(xr + 0);
      *(float4*)&av[4]  = *(const float4*)(xr + 4);
      *(float4*)&av[8]  = *(const float4*)(xr + 16);
      *(float4*)&av[12] = *(const float4*)(xr + 20);
      const v16bf afrag = pack16_bf(av);
#pragma unroll
      for (int nt = 0; nt < NT; ++nt) {
        const int nrow = (wave * NT + nt) * 16 + bn;
        const float* wr = W + (size_t)nrow * IC_N + ks * 32 + bkoff;
        __builtin_prefetch(wr + 32, 0, 1);
        float wv[16];
        *(float4*)&wv[0]  = *(const float4*)(wr + 0);
        *(float4*)&wv[4]  = *(const float4*)(wr + 4);
        *(float4*)&wv[8]  = *(const float4*)(wr + 16);
        *(float4*)&wv[12] = *(const float4*)(wr + 20);
        acc[nt] = wmma_bf16f32(afrag, pack16_bf(wv), acc[nt]);
      }
    }
    __syncthreads();
    const float* bias = fc1_b + (size_t)t * HID;
#pragma unroll
    for (int nt = 0; nt < NT; ++nt) {
      const int nglob = (wave * NT + nt) * 16 + bn;
      const float bv = bias[nglob];
#pragma unroll
      for (int j = 0; j < 8; ++j)
        hsh[(mbase + j) * HID + nglob] = (__bf16)silu_f(acc[nt][j] + bv);
    }
    __syncthreads();
  }

  // ---------------- residual blocks: h += silu(GN(h) @ W.T + b) ----------------
  for (int l = 0; l < L_N; ++l) {
    // per-(row,group) mean / rstd
    for (int p = tid; p < 16 * 32; p += BDIM) {
      const int row = p >> 5, g = p & 31;
      const __bf16* hp = &hsh[row * HID + g * GS];
      float s = 0.0f, ss = 0.0f;
#pragma unroll 4
      for (int c = 0; c < GS; ++c) {
        const float v = (float)hp[c];
        s += v;
        ss += v * v;
      }
      const float mean = s * (1.0f / GS);
      const float var  = ss * (1.0f / GS) - mean * mean;
      smean[p] = mean;
      srstd[p] = rsqrtf(var + EPS_F);
    }
    __syncthreads();

    const float* gamma = gn_g + ((size_t)t * L_N + l) * HID;
    const float* beta  = gn_b + ((size_t)t * L_N + l) * HID;
    const float* W     = blk_w + ((size_t)t * L_N + l) * HID * HID;
    const float* bias  = blk_b + ((size_t)t * L_N + l) * HID;

#pragma unroll
    for (int nt = 0; nt < NT; ++nt)
#pragma unroll
      for (int j = 0; j < 8; ++j) acc[nt][j] = 0.0f;

    for (int ks = 0; ks < KS_BLK; ++ks) {
      const int c0 = ks * 32 + akoff;
      const int c1 = c0 + 16;
      const v8bf lo = *(const v8bf*)&hsh[am * HID + c0];  // ds_load_b128
      const v8bf hi = *(const v8bf*)&hsh[am * HID + c1];
      float ga[16], bt[16];
      *(float4*)&ga[0]  = *(const float4*)(gamma + c0);
      *(float4*)&ga[4]  = *(const float4*)(gamma + c0 + 4);
      *(float4*)&ga[8]  = *(const float4*)(gamma + c1);
      *(float4*)&ga[12] = *(const float4*)(gamma + c1 + 4);
      *(float4*)&bt[0]  = *(const float4*)(beta + c0);
      *(float4*)&bt[4]  = *(const float4*)(beta + c0 + 4);
      *(float4*)&bt[8]  = *(const float4*)(beta + c1);
      *(float4*)&bt[12] = *(const float4*)(beta + c1 + 4);
      float av[16];
#pragma unroll
      for (int e = 0; e < 8; ++e) {
        const int c = c0 + e;
        const int g = c / GS;
        av[e] = ((float)lo[e] - smean[am * 32 + g]) * srstd[am * 32 + g] * ga[e] + bt[e];
      }
#pragma unroll
      for (int e = 0; e < 8; ++e) {
        const int c = c1 + e;
        const int g = c / GS;
        av[8 + e] =
            ((float)hi[e] - smean[am * 32 + g]) * srstd[am * 32 + g] * ga[8 + e] + bt[8 + e];
      }
      const v16bf afrag = pack16_bf(av);
#pragma unroll
      for (int nt = 0; nt < NT; ++nt) {
        const int nrow = (wave * NT + nt) * 16 + bn;
        const float* wr = W + (size_t)nrow * HID + ks * 32 + bkoff;
        __builtin_prefetch(wr + 32, 0, 1);
        float wv[16];
        *(float4*)&wv[0]  = *(const float4*)(wr + 0);
        *(float4*)&wv[4]  = *(const float4*)(wr + 4);
        *(float4*)&wv[8]  = *(const float4*)(wr + 16);
        *(float4*)&wv[12] = *(const float4*)(wr + 20);
        acc[nt] = wmma_bf16f32(afrag, pack16_bf(wv), acc[nt]);
      }
    }
    __syncthreads();  // all A-reads of hsh done before residual update
#pragma unroll
    for (int nt = 0; nt < NT; ++nt) {
      const int nglob = (wave * NT + nt) * 16 + bn;
      const float bv = bias[nglob];
#pragma unroll
      for (int j = 0; j < 8; ++j) {
        const int m = mbase + j;
        const float y  = silu_f(acc[nt][j] + bv);
        const float hn = (float)hsh[m * HID + nglob] + y;
        hsh[m * HID + nglob] = (__bf16)hn;
      }
    }
    __syncthreads();
  }

  // ---------------- out-proj + scatter: out = h @ out_w.T + out_b ----------------
  {
    const float* W    = out_w + (size_t)t * HID * HID;
    const float* bias = out_b + (size_t)t * HID;
#pragma unroll
    for (int nt = 0; nt < NT; ++nt)
#pragma unroll
      for (int j = 0; j < 8; ++j) acc[nt][j] = 0.0f;

    for (int ks = 0; ks < KS_BLK; ++ks) {
      const int c0 = ks * 32 + akoff;
      const v8bf lo = *(const v8bf*)&hsh[am * HID + c0];
      const v8bf hi = *(const v8bf*)&hsh[am * HID + c0 + 16];
      v16bf afrag;
#pragma unroll
      for (int e = 0; e < 8; ++e) {
        afrag[e]     = lo[e];
        afrag[8 + e] = hi[e];
      }
#pragma unroll
      for (int nt = 0; nt < NT; ++nt) {
        const int nrow = (wave * NT + nt) * 16 + bn;
        const float* wr = W + (size_t)nrow * HID + ks * 32 + bkoff;
        __builtin_prefetch(wr + 32, 0, 1);
        float wv[16];
        *(float4*)&wv[0]  = *(const float4*)(wr + 0);
        *(float4*)&wv[4]  = *(const float4*)(wr + 4);
        *(float4*)&wv[8]  = *(const float4*)(wr + 16);
        *(float4*)&wv[12] = *(const float4*)(wr + 20);
        acc[nt] = wmma_bf16f32(afrag, pack16_bf(wv), acc[nt]);
      }
    }

    // Final scatter: block B_t reversed for t=0..3, plus B_4 forward for t=3.
#pragma unroll
    for (int nt = 0; nt < NT; ++nt) {
      const int nglob = (wave * NT + nt) * 16 + bn;
      const float bv = bias[nglob];
#pragma unroll
      for (int j = 0; j < 8; ++j) {
        const int m = mbase + j;
        const float v = acc[nt][j] + bv;
        const int r = r0 + m;            // global row in [0,128)
        const int b = r >> 4;            // batch
        const int s = r & 15;            // seq index
        const int np = npos[b];
        if (s < np) {
          out[((size_t)b * S_N + (size_t)(S_N - 1 - t * np - s)) * HID + nglob] = v;
          if (t == T_N - 1)
            out[((size_t)b * S_N + (size_t)(S_N - 5 * np + s)) * HID + nglob] = v;
        }
      }
    }
  }
}

// ---------------------------------------------------------------------------
extern "C" void kernel_launch(void* const* d_in, const int* in_sizes, int n_in,
                              void* d_out, int out_size, void* d_ws, size_t ws_size,
                              hipStream_t stream) {
  const float* ehs0  = (const float*)d_in[0];   // [8,4096,768]
  const float* ehs1  = (const float*)d_in[1];   // [8,4096,1280]
  const float* ehs2  = (const float*)d_in[2];   // [8,2816]
  const float* extra = (const float*)d_in[3];   // [8,16,704]

  const float* te1_fc1_w = (const float*)d_in[4];
  const float* te1_fc1_b = (const float*)d_in[5];
  const float* te1_gn_g  = (const float*)d_in[6];
  const float* te1_gn_b  = (const float*)d_in[7];
  const float* te1_blk_w = (const float*)d_in[8];
  const float* te1_blk_b = (const float*)d_in[9];
  const float* te1_out_w = (const float*)d_in[10];
  const float* te1_out_b = (const float*)d_in[11];

  const float* te2_fc1_w = (const float*)d_in[12];
  const float* te2_fc1_b = (const float*)d_in[13];
  const float* te2_gn_g  = (const float*)d_in[14];
  const float* te2_gn_b  = (const float*)d_in[15];
  const float* te2_blk_w = (const float*)d_in[16];
  const float* te2_blk_b = (const float*)d_in[17];
  const float* te2_out_w = (const float*)d_in[18];
  const float* te2_out_b = (const float*)d_in[19];

  float* out0 = (float*)d_out;                                   // ehs0' region
  float* out1 = out0 + (size_t)B_N * S_N * H1_N;                 // ehs1' region
  float* out2 = out1 + (size_t)B_N * S_N * H2_N;                 // ehs2' region
  int* npos   = (int*)d_ws;

  // 1) baseline copies: big regions through the async Global<->LDS engine
  const long n0 = (long)B_N * S_N * H1_N / 4;
  const long n1 = (long)B_N * S_N * H2_N / 4;
  const long n2 = (long)B_N * 2816 / 4;
  copy_async_kernel<<<4096, BDIM, 0, stream>>>((const float4*)ehs0, (float4*)out0, n0);
  copy_async_kernel<<<4096, BDIM, 0, stream>>>((const float4*)ehs1, (float4*)out1, n1);
  copy4_kernel<<<32, BDIM, 0, stream>>>((const float4*)ehs2, (float4*)out2, n2);

  // 2) data-dependent valid-token counts
  npos_kernel<<<B_N, BDIM, 0, stream>>>(extra, npos);

  // 3) fused WMMA ResMLPs + scatter (grid: 8 row-tiles x T=4 param sets)
  dim3 grid(8, T_N);
  resmlp_wmma_kernel<H1_N><<<grid, BDIM, 0, stream>>>(
      extra, te1_fc1_w, te1_fc1_b, te1_gn_g, te1_gn_b, te1_blk_w, te1_blk_b,
      te1_out_w, te1_out_b, npos, out0);
  resmlp_wmma_kernel<H2_N><<<grid, BDIM, 0, stream>>>(
      extra, te2_fc1_w, te2_fc1_b, te2_gn_g, te2_gn_b, te2_blk_w, te2_blk_b,
      te2_out_w, te2_out_b, npos, out1);
}
